// Attention_18442589569444
// MI455X (gfx1250) — compile-verified, matrix operand order checked by reading, not run on hardware
//
#include <hip/hip_runtime.h>

// ---------------------------------------------------------------------------
// Types for CDNA5 WMMA (wave32, 16x16x32 bf16 -> f32)
// ---------------------------------------------------------------------------
typedef __bf16 bf16;
typedef __attribute__((ext_vector_type(16))) __bf16 v16bf;
typedef __attribute__((ext_vector_type(8)))  __bf16 bf16x8;
typedef __attribute__((ext_vector_type(4)))  __bf16 bf16x4;
typedef __attribute__((ext_vector_type(8)))  float  v8f;

constexpr int B_  = 2;
constexpr int T_  = 2048;
constexpr int C_  = 2048;
constexpr int NH_ = 16;
constexpr int H_  = 128;

union Frag {
  v16bf  v;
  bf16x8 h[2];
};

__device__ __forceinline__ v8f wmma_bf16(const Frag& a, const Frag& b, v8f c) {
  // D = A(16x32) * B(32x16) + C, fp32 accum
  return __builtin_amdgcn_wmma_f32_16x16x32_bf16(
      /*neg_a=*/false, a.v, /*neg_b=*/false, b.v,
      /*c_mod=*/(short)0, c, /*reuse_a=*/false, /*reuse_b=*/false);
}

// Async global->LDS copy, 16 bytes per lane, tracked by ASYNCcnt (CDNA5 §08).
// lds: 32-bit LDS byte address; gaddr: 64-bit global address.
#define ASYNC_CP16(lds, gaddr, imm)                                          \
  asm volatile("global_load_async_to_lds_b128 %0, %1, off offset:" #imm     \
               :: "v"(lds), "v"(gaddr) : "memory")
#define WAIT_ASYNC() asm volatile("s_wait_asynccnt 0x0" ::: "memory")

__device__ __forceinline__ unsigned lds_addr_of(const void* p) {
  // Low 32 bits of a flat LDS address are the LDS byte offset (ISA §10.2).
  return (unsigned)(size_t)p;
}

// ---------------------------------------------------------------------------
// fp32 -> bf16 conversion (vectorized x4; all sizes here are multiples of 4)
// ---------------------------------------------------------------------------
__global__ __launch_bounds__(256) void cvt_f32_bf16(
    const float* __restrict__ in, bf16* __restrict__ out, int n) {
  const int i = (blockIdx.x * 256 + threadIdx.x) * 4;
  if (i + 3 < n) {
    const float4 v = *(const float4*)(in + i);
    bf16x4 o;
    o[0] = (bf16)v.x; o[1] = (bf16)v.y; o[2] = (bf16)v.z; o[3] = (bf16)v.w;
    *(bf16x4*)(out + i) = o;
  }
}

// ---------------------------------------------------------------------------
// GEMM: Out[m][n] = sum_k A[m][k] * W[n][k]   (i.e. A @ W^T)
// Block tile 128x128, K-step 32, 256 threads = 8 waves, each wave 32x64.
// Double-buffered LDS staged with global_load_async_to_lds_b128.
// ---------------------------------------------------------------------------
template <bool OUTF32>
__global__ __launch_bounds__(256) void gemm_tn(
    const bf16* __restrict__ A, const bf16* __restrict__ W,
    void* __restrict__ OutP, const int M, const int N, const int K) {
  __shared__ __align__(16) bf16 As[2][128][32];   // 2 x 8 KB
  __shared__ __align__(16) bf16 Ws[2][128][32];   // 2 x 8 KB

  const int tid  = threadIdx.x;
  const int lane = tid & 31;
  const int wave = tid >> 5;
  const int wm   = wave >> 1;          // 0..3  (32-row strip)
  const int wn   = wave & 1;           // 0..1  (64-col strip)
  const int lr   = lane & 15;
  const int hi8  = (lane >> 4) * 8;    // A/B frag K sub-offset per ISA layout
  const size_t bm = (size_t)blockIdx.x * 128;
  const size_t bn = (size_t)blockIdx.y * 128;

  v8f acc[2][4];
#pragma unroll
  for (int i = 0; i < 2; ++i)
#pragma unroll
    for (int j = 0; j < 4; ++j)
#pragma unroll
      for (int e = 0; e < 8; ++e) acc[i][j][e] = 0.f;

  const int srow = tid >> 1;           // 0..127 staging row
  const int scol = (tid & 1) * 16;     // 0 or 16
  const bf16* ap = A + (bm + srow) * (size_t)K + scol;
  const bf16* wp = W + (bn + srow) * (size_t)K + scol;

  // Async-stage one 128x32 slab of A and W into buffer `buf`.
  auto stage = [&](int buf, int k0) {
    const unsigned la = lds_addr_of(&As[buf][srow][scol]);
    const unsigned lw = lds_addr_of(&Ws[buf][srow][scol]);
    const unsigned long long ga = (unsigned long long)(size_t)(ap + k0);
    const unsigned long long gw = (unsigned long long)(size_t)(wp + k0);
    ASYNC_CP16(la, ga, 0);
    ASYNC_CP16(la, ga, 16);
    ASYNC_CP16(lw, gw, 0);
    ASYNC_CP16(lw, gw, 16);
  };

  // WMMA over one staged slab.
  auto compute = [&](int buf) {
    Frag af[2], wf[4];
#pragma unroll
    for (int i = 0; i < 2; ++i) {
      const int r = wm * 32 + i * 16 + lr;
      af[i].h[0] = *(const bf16x8*)&As[buf][r][hi8];
      af[i].h[1] = *(const bf16x8*)&As[buf][r][hi8 + 16];
    }
#pragma unroll
    for (int j = 0; j < 4; ++j) {
      const int c = wn * 64 + j * 16 + lr;
      wf[j].h[0] = *(const bf16x8*)&Ws[buf][c][hi8];
      wf[j].h[1] = *(const bf16x8*)&Ws[buf][c][hi8 + 16];
    }
#pragma unroll
    for (int i = 0; i < 2; ++i)
#pragma unroll
      for (int j = 0; j < 4; ++j)
        acc[i][j] = wmma_bf16(af[i], wf[j], acc[i][j]);
  };

  stage(0, 0);
  WAIT_ASYNC();
  __syncthreads();

  for (int k0 = 0; k0 < K; k0 += 64) {
    stage(1, k0 + 32);                 // prefetch next slab while computing
    compute(0);
    WAIT_ASYNC();
    __syncthreads();

    if (k0 + 64 < K) stage(0, k0 + 64);
    compute(1);
    WAIT_ASYNC();
    __syncthreads();
  }

  const int moff = (lane >> 4) * 8;    // C/D layout: lanes 16-31 hold rows 8..15
#pragma unroll
  for (int i = 0; i < 2; ++i)
#pragma unroll
    for (int j = 0; j < 4; ++j)
#pragma unroll
      for (int r = 0; r < 8; ++r) {
        const size_t gm = bm + wm * 32 + i * 16 + moff + r;
        const size_t gn = bn + wn * 64 + j * 16 + lr;
        if constexpr (OUTF32)
          ((float*)OutP)[gm * N + gn] = acc[i][j][r];
        else
          ((bf16*)OutP)[gm * N + gn] = (bf16)acc[i][j][r];
      }
}

// ---------------------------------------------------------------------------
// RoPE + RMSNorm for Q and K. One block per (b,t,h) row, 128 threads.
// ---------------------------------------------------------------------------
__global__ __launch_bounds__(128) void rope_rms_kernel(
    const bf16* __restrict__ Qr, const bf16* __restrict__ Kr,
    const float* __restrict__ S, const float* __restrict__ Cs,
    const float* __restrict__ qw, const float* __restrict__ kw,
    bf16* __restrict__ Qo, bf16* __restrict__ Ko) {
  __shared__ float red[4];
  const int idx = blockIdx.x;
  const int h = idx % NH_;
  const int t = (idx / NH_) % T_;
  const int b = idx / (NH_ * T_);
  const int d = threadIdx.x;
  const size_t row = ((size_t)b * T_ + t) * C_ + (size_t)h * H_;
  const float sn = S[(size_t)t * (H_ / 2) + (d & 63)];
  const float cs = Cs[(size_t)t * (H_ / 2) + (d & 63)];

  {  // Q
    const float x = (float)Qr[row + d];
    const float y = (float)Qr[row + (d ^ 64)];
    const float v = (d < 64) ? (x * cs - y * sn) : (y * sn + x * cs);
    float ss = v * v;
#pragma unroll
    for (int m = 16; m >= 1; m >>= 1) ss += __shfl_xor(ss, m);
    if ((threadIdx.x & 31) == 0) red[threadIdx.x >> 5] = ss;
    __syncthreads();
    const float inv =
        rsqrtf((red[0] + red[1] + red[2] + red[3]) * (1.f / H_) + 1.1920929e-7f);
    Qo[row + d] = (bf16)(v * inv * qw[d]);
  }
  __syncthreads();
  {  // K
    const float x = (float)Kr[row + d];
    const float y = (float)Kr[row + (d ^ 64)];
    const float v = (d < 64) ? (x * cs - y * sn) : (y * sn + x * cs);
    float ss = v * v;
#pragma unroll
    for (int m = 16; m >= 1; m >>= 1) ss += __shfl_xor(ss, m);
    if ((threadIdx.x & 31) == 0) red[threadIdx.x >> 5] = ss;
    __syncthreads();
    const float inv =
        rsqrtf((red[0] + red[1] + red[2] + red[3]) * (1.f / H_) + 1.1920929e-7f);
    Ko[row + d] = (bf16)(v * inv * kw[d]);
  }
}

// ---------------------------------------------------------------------------
// Flash attention with WMMA. 128 queries/block, 64-key tiles, causal.
// 8 waves, each owns 16 query rows; S = Q K^T and O += P V via WMMA.
// K tile staged with async global->LDS copies; V transposed through VGPRs.
// ---------------------------------------------------------------------------
__global__ __launch_bounds__(256) void flash_attn_kernel(
    const bf16* __restrict__ Q, const bf16* __restrict__ K,
    const bf16* __restrict__ V, bf16* __restrict__ O) {
  __shared__ __align__(16) bf16 Ks[64][H_];      // 16 KB  [key][d]
  __shared__ __align__(16) bf16 Vt[H_][64];      // 16 KB  [d][key] (transposed)
  __shared__ __align__(16) bf16 Ps[8][16][64];   // 16 KB  per-wave P strip

  const int tid  = threadIdx.x;
  const int lane = tid & 31;
  const int wave = tid >> 5;
  const int lr   = lane & 15;
  const int hi8  = (lane >> 4) * 8;   // frag K sub-offset / C-row offset
  const int moff = hi8;

  const int bh = blockIdx.y;
  const int b  = bh / NH_;
  const int h  = bh % NH_;
  const int q0 = blockIdx.x * 128;
  const size_t base = ((size_t)b * T_) * C_ + (size_t)h * H_;

  // Q fragments for this wave's 16 rows (A-layout, 4 chunks over d)
  Frag qf[4];
  {
    const bf16* qp = Q + base + (size_t)(q0 + wave * 16 + lr) * C_;
#pragma unroll
    for (int kc = 0; kc < 4; ++kc) {
      qf[kc].h[0] = *(const bf16x8*)(qp + kc * 32 + hi8);
      qf[kc].h[1] = *(const bf16x8*)(qp + kc * 32 + hi8 + 16);
    }
  }

  v8f o[8];
#pragma unroll
  for (int j = 0; j < 8; ++j)
#pragma unroll
    for (int e = 0; e < 8; ++e) o[j][e] = 0.f;
  float rm[8], rl[8];
#pragma unroll
  for (int r = 0; r < 8; ++r) { rm[r] = -1e30f; rl[r] = 0.f; }

  const int ktiles = blockIdx.x * 2 + 2;   // causal: keys up to q0+127
  const int srow = tid >> 2;               // staging: key row 0..63
  const int scol = (tid & 3) * 32;         // staging: d base

  for (int kt = 0; kt < ktiles; ++kt) {
    const int k0 = kt * 64;
    __syncthreads();
    {
      // K tile: async global->LDS, 64 bytes per lane
      const unsigned lk = lds_addr_of(&Ks[srow][scol]);
      const unsigned long long gk =
          (unsigned long long)(size_t)(K + base + (size_t)(k0 + srow) * C_ + scol);
      ASYNC_CP16(lk, gk, 0);
      ASYNC_CP16(lk, gk, 16);
      ASYNC_CP16(lk, gk, 32);
      ASYNC_CP16(lk, gk, 48);

      // V tile: load through VGPRs and scatter transposed into LDS
      const bf16* vp = V + base + (size_t)(k0 + srow) * C_ + scol;
      bf16x8 vv[4];
#pragma unroll
      for (int u = 0; u < 4; ++u) vv[u] = *(const bf16x8*)(vp + u * 8);
#pragma unroll
      for (int u = 0; u < 4; ++u)
#pragma unroll
        for (int e = 0; e < 8; ++e) Vt[scol + u * 8 + e][srow] = vv[u][e];

      WAIT_ASYNC();
    }
    __syncthreads();

    // S = Q K^T : 4 tiles of 16x16, K-depth 128 => 16 WMMAs
    v8f s[4];
#pragma unroll
    for (int j = 0; j < 4; ++j)
#pragma unroll
      for (int e = 0; e < 8; ++e) s[j][e] = 0.f;
#pragma unroll
    for (int j = 0; j < 4; ++j) {
      Frag kf[4];
#pragma unroll
      for (int kc = 0; kc < 4; ++kc) {
        kf[kc].h[0] = *(const bf16x8*)&Ks[j * 16 + lr][kc * 32 + hi8];
        kf[kc].h[1] = *(const bf16x8*)&Ks[j * 16 + lr][kc * 32 + hi8 + 16];
      }
#pragma unroll
      for (int kc = 0; kc < 4; ++kc) s[j] = wmma_bf16(qf[kc], kf[kc], s[j]);
    }

    // Online softmax (per-row stats duplicated across 16-lane groups)
    const float scale = 0.08838834764831845f;  // 1/sqrt(128)
    const int qrow0 = q0 + wave * 16 + moff;
#pragma unroll
    for (int r = 0; r < 8; ++r) {
      const int gq = qrow0 + r;
      float mx = -1e30f;
#pragma unroll
      for (int j = 0; j < 4; ++j) {
        const int gk = k0 + j * 16 + lr;
        float v = s[j][r] * scale;
        v = (gk <= gq) ? v : -1e30f;   // causal mask (select, not branch)
        s[j][r] = v;
        mx = fmaxf(mx, v);
      }
#pragma unroll
      for (int m = 8; m >= 1; m >>= 1) mx = fmaxf(mx, __shfl_xor(mx, m));
      const float mnew  = fmaxf(rm[r], mx);
      const float alpha = __expf(rm[r] - mnew);
      float rs = 0.f;
#pragma unroll
      for (int j = 0; j < 4; ++j) {
        const float p = __expf(s[j][r] - mnew);
        rs += p;
        Ps[wave][moff + r][j * 16 + lr] = (bf16)p;
      }
#pragma unroll
      for (int m = 8; m >= 1; m >>= 1) rs += __shfl_xor(rs, m);
      rl[r] = rl[r] * alpha + rs;
      rm[r] = mnew;
#pragma unroll
      for (int jd = 0; jd < 8; ++jd) o[jd][r] *= alpha;
    }

    // O += P V : P from per-wave LDS (same-wave DS ordering), 16 WMMAs
    Frag pf[2];
#pragma unroll
    for (int kc = 0; kc < 2; ++kc) {
      pf[kc].h[0] = *(const bf16x8*)&Ps[wave][lr][kc * 32 + hi8];
      pf[kc].h[1] = *(const bf16x8*)&Ps[wave][lr][kc * 32 + hi8 + 16];
    }
#pragma unroll
    for (int jd = 0; jd < 8; ++jd) {
      Frag vf[2];
#pragma unroll
      for (int kc = 0; kc < 2; ++kc) {
        vf[kc].h[0] = *(const bf16x8*)&Vt[jd * 16 + lr][kc * 32 + hi8];
        vf[kc].h[1] = *(const bf16x8*)&Vt[jd * 16 + lr][kc * 32 + hi8 + 16];
      }
#pragma unroll
      for (int kc = 0; kc < 2; ++kc) o[jd] = wmma_bf16(pf[kc], vf[kc], o[jd]);
    }
  }

  // Normalize and write O (bf16, [B,T,C] with head offset in base)
#pragma unroll
  for (int r = 0; r < 8; ++r) {
    const float inv = 1.f / rl[r];
    bf16* op = O + base + (size_t)(q0 + wave * 16 + moff + r) * C_;
#pragma unroll
    for (int jd = 0; jd < 8; ++jd)
      op[jd * 16 + lr] = (bf16)(o[jd][r] * inv);
  }
}

// ---------------------------------------------------------------------------
// Orchestration. Workspace layout (peak 96 MiB, regions reused):
//   [0,16)   xb  -> later qr
//   [16,24)  Wq bf16 -> later (with [24,32)) kr
//   [24,32)  Wk bf16
//   [32,40)  Wv bf16
//   [40,48)  Wo bf16 (live until final GEMM)
//   [48,64)  Q raw -> later attn output Ob
//   [64,80)  K raw
//   [80,96)  V raw (live through flash attention)
// ---------------------------------------------------------------------------
extern "C" void kernel_launch(void* const* d_in, const int* in_sizes, int n_in,
                              void* d_out, int out_size, void* d_ws,
                              size_t ws_size, hipStream_t stream) {
  const float* x  = (const float*)d_in[0];
  const float* sn = (const float*)d_in[1];
  const float* cs = (const float*)d_in[2];
  const float* Wq = (const float*)d_in[3];
  const float* Wk = (const float*)d_in[4];
  const float* Wv = (const float*)d_in[5];
  const float* Wo = (const float*)d_in[6];
  const float* qw = (const float*)d_in[7];
  const float* kw = (const float*)d_in[8];
  float* out = (float*)d_out;

  char* w = (char*)d_ws;
  const size_t MiB = 1ull << 20;
  bf16* xb   = (bf16*)(w + 0);
  bf16* wqb  = (bf16*)(w + 16 * MiB);
  bf16* wkb  = (bf16*)(w + 24 * MiB);
  bf16* wvb  = (bf16*)(w + 32 * MiB);
  bf16* wob  = (bf16*)(w + 40 * MiB);
  bf16* qraw = (bf16*)(w + 48 * MiB);
  bf16* kraw = (bf16*)(w + 64 * MiB);
  bf16* vraw = (bf16*)(w + 80 * MiB);
  bf16* qr   = (bf16*)(w + 0);        // reuse xb region
  bf16* kr   = (bf16*)(w + 16 * MiB); // reuse wqb/wkb region
  bf16* ob   = (bf16*)(w + 48 * MiB); // reuse qraw region

  const int NX = B_ * T_ * C_;  // 8388608
  const int NW = C_ * C_;       // 4194304
  cvt_f32_bf16<<<NX / 4 / 256, 256, 0, stream>>>(x, xb, NX);
  cvt_f32_bf16<<<NW / 4 / 256, 256, 0, stream>>>(Wq, wqb, NW);
  cvt_f32_bf16<<<NW / 4 / 256, 256, 0, stream>>>(Wk, wkb, NW);
  cvt_f32_bf16<<<NW / 4 / 256, 256, 0, stream>>>(Wv, wvb, NW);
  cvt_f32_bf16<<<NW / 4 / 256, 256, 0, stream>>>(Wo, wob, NW);

  dim3 gg(B_ * T_ / 128, C_ / 128);  // (32, 16)
  gemm_tn<false><<<gg, 256, 0, stream>>>(xb, wqb, qraw, B_ * T_, C_, C_);
  gemm_tn<false><<<gg, 256, 0, stream>>>(xb, wkb, kraw, B_ * T_, C_, C_);
  gemm_tn<false><<<gg, 256, 0, stream>>>(xb, wvb, vraw, B_ * T_, C_, C_);

  rope_rms_kernel<<<B_ * T_ * NH_, 128, 0, stream>>>(qraw, kraw, sn, cs, qw, kw,
                                                     qr, kr);

  dim3 fg(T_ / 128, B_ * NH_);  // (16, 32)
  flash_attn_kernel<<<fg, 256, 0, stream>>>(qr, kr, vraw, ob);

  gemm_tn<true><<<gg, 256, 0, stream>>>(ob, wob, out, B_ * T_, C_, C_);

  (void)in_sizes; (void)n_in; (void)out_size; (void)ws_size;
}
